// ContrastiveLoss_5497558139284
// MI455X (gfx1250) — matrix operand order repaired and checked by reading, not compiled, not run on hardware
//
#include <hip/hip_runtime.h>

// Fused contrastive loss for MI455X (gfx1250), wave32 + bf16 WMMA + async LDS copy.
// out[(i*T+t)*T+s] = -( log1p(sum_{j!=i} exp(<V_i[t],A_j[s]>/(|V_i|_F*cnA[j,s])))
//                     + log1p(sum_{j!=i} exp(<A_i[t],V_j[s]>/(|A_i|_F*cnV[j,s]))) )
// B=32, T=D=256. Compute-bound (~69 GFLOP GEMM + 134M exp); data fits in L2.
// Pipeline: (1) convert f32 -> bf16 once, (2) norms, (3) fused batched GEMM with
// double-buffered GLOBAL_LOAD_ASYNC_TO_LDS staging overlapping v_wmma bf16 math.

typedef __attribute__((ext_vector_type(16))) __bf16 v16bf;
typedef __attribute__((ext_vector_type(8)))  float  v8f;

#define EPS 1e-18f
#define NH  (32 * 256 * 256)  // elements per tensor

#if defined(__has_builtin)
#if __has_builtin(__builtin_amdgcn_global_load_async_to_lds_b128)
#define USE_ASYNC_LDS 1
#endif
#endif

// Builtin takes pointers to 16-byte int vectors in AS(1)/AS(3).
typedef int v4i __attribute__((vector_size(16)));
typedef __attribute__((address_space(1))) v4i gv4i_t;
typedef __attribute__((address_space(3))) v4i lv4i_t;

// 16-byte global -> LDS copy; async (ASYNCcnt) when the gfx1250 builtin exists.
// Flat->AS3: truncate flat address to 32 bits (LDS offset per aperture rules).
__device__ __forceinline__ void cp16(unsigned short* ldst, const unsigned short* gsrc) {
#ifdef USE_ASYNC_LDS
  __builtin_amdgcn_global_load_async_to_lds_b128(
      (gv4i_t*)(unsigned long long)gsrc,
      (lv4i_t*)(unsigned int)(unsigned long long)ldst, 0, 0);
#else
  *(uint4*)ldst = *(const uint4*)gsrc;
#endif
}

__device__ __forceinline__ void wait_async() {
#ifdef USE_ASYNC_LDS
#if __has_builtin(__builtin_amdgcn_s_wait_asynccnt)
  __builtin_amdgcn_s_wait_asynccnt(0);
#else
  asm volatile("s_wait_asynccnt 0x0" ::: "memory");
#endif
#endif
}

// round-to-nearest-even f32 -> bf16, two at a time packed into a u32
__device__ __forceinline__ unsigned int f2bf2(float lo, float hi) {
  unsigned int a = __builtin_bit_cast(unsigned int, lo);
  unsigned int b = __builtin_bit_cast(unsigned int, hi);
  a = (a + 0x7FFFu + ((a >> 16) & 1u)) >> 16;
  b = (b + 0x7FFFu + ((b >> 16) & 1u)) & 0xFFFF0000u;
  return a | b;
}

__device__ __forceinline__ uint4 pack8(float4 x, float4 y) {
  uint4 r;
  r.x = f2bf2(x.x, x.y);
  r.y = f2bf2(x.z, x.w);
  r.z = f2bf2(y.x, y.y);
  r.w = f2bf2(y.z, y.w);
  return r;
}

union FragU { uint4 u[2]; v16bf v; };
__device__ __forceinline__ v16bf mkfrag(uint4 a, uint4 b) {
  FragU f; f.u[0] = a; f.u[1] = b; return f.v;
}

// ---------------------------------------------------------------------------
// Kernel 0: one-shot f32 -> bf16 conversion of both tensors into workspace.
// ---------------------------------------------------------------------------
__global__ __launch_bounds__(256) void convert_kernel(
    const float* __restrict__ V, const float* __restrict__ A,
    unsigned short* __restrict__ bfV, unsigned short* __restrict__ bfA) {
  const size_t e = ((size_t)blockIdx.x * 256 + threadIdx.x) * 8;
  const float4* gv = (const float4*)(V + e);
  *(uint4*)(bfV + e) = pack8(gv[0], gv[1]);
  const float4* ga = (const float4*)(A + e);
  *(uint4*)(bfA + e) = pack8(ga[0], ga[1]);
}

// ---------------------------------------------------------------------------
// Kernel 1: per-column norms [B,256] and Frobenius norms [B] for both tensors.
// nrm layout (floats): [0,8192)=colnormV [8192,16384)=colnormA
//                      [16384,16416)=frobV [16416,16448)=frobA
// ---------------------------------------------------------------------------
__global__ __launch_bounds__(256) void norms_kernel(
    const float* __restrict__ V, const float* __restrict__ A,
    float* __restrict__ nrm) {
  const int b      = blockIdx.x;
  const int tensor = b >> 5;
  const int i      = b & 31;
  const int s      = threadIdx.x;
  const float* __restrict__ X = tensor ? A : V;
  const float* base = X + (size_t)i * 65536 + s;
  float sum = 0.f;
#pragma unroll 4
  for (int t = 0; t < 256; ++t) {
    float x = base[(size_t)t * 256];
    sum += x * x;
  }
  nrm[tensor * 8192 + i * 256 + s] = sqrtf(sum + EPS);

  __shared__ float red[256];
  red[s] = sum;
  __syncthreads();
  for (int off = 128; off > 0; off >>= 1) {
    if (s < off) red[s] += red[s + off];
    __syncthreads();
  }
  if (s == 0) nrm[16384 + tensor * 32 + i] = sqrtf(red[0] + EPS);
}

// ---------------------------------------------------------------------------
// Kernel 2: fused batched-GEMM + exp-sum + log1p on bf16 data.
// Grid: 32 anchors * 16 tiles (64x64 of the [t,s] plane). 8 waves per block;
// wave w -> rows [w/2*16,+16), cols [(w&1)*32,+32) = 2 WMMA C tiles.
// Double-buffered async K staging overlaps copies with WMMA; 1 barrier/step.
// ---------------------------------------------------------------------------
__global__ __launch_bounds__(256) void loss_kernel(
    const unsigned short* __restrict__ bfV, const unsigned short* __restrict__ bfA,
    const float* __restrict__ nrm, float* __restrict__ out) {
  __shared__ alignas(16) unsigned short lQ[64 * 256];     // query strip (32KB)
  __shared__ alignas(16) unsigned short lK[2][64 * 32];   // key tiles, 2x4KB
  __shared__ float lS[32 * 64];                           // 1/(frob*colnorm) (8KB)

  const int i  = blockIdx.x >> 4;
  const int tl = blockIdx.x & 15;
  const int t0 = (tl >> 2) * 64;
  const int s0 = (tl & 3) * 64;

  const int tid  = threadIdx.x;
  const int lane = tid & 31;
  const int wid  = tid >> 5;
  const int m0   = (wid >> 1) * 16;
  const int wcol = wid & 1;
  const int lh   = lane & 15;
  const int hi   = lane >> 4;

  const int ldrow = tid >> 2;        // 0..63   (key-tile staging)
  const int ldcol = (tid & 3) << 3;  // 0,8,16,24

  const v8f vzero = {};
  float dsum[2][2][8];

  for (int dir = 0; dir < 2; ++dir) {
    const unsigned short* __restrict__ Qb = dir ? bfA : bfV;   // query rows (t)
    const unsigned short* __restrict__ Kb = dir ? bfV : bfA;   // key rows (s)
    const float* __restrict__ cn = dir ? nrm : (nrm + 8192);   // colnorm of keys
    const float frob = nrm[16384 + dir * 32 + i];

    __syncthreads();  // previous direction fully done with lQ/lK/lS

    // Stage query strip (async) and precomputed inverse scales (VALU, once).
    {
      const int qrow = tid >> 2;
      const int qcol = (tid & 3) << 6;
      const unsigned short* g = Qb + ((size_t)(i * 256 + t0 + qrow)) * 256 + qcol;
#pragma unroll
      for (int c = 0; c < 8; ++c)
        cp16(&lQ[qrow * 256 + qcol + c * 8], g + c * 8);
    }
    {
      const float rf = 1.f / frob;
#pragma unroll
      for (int e = 0; e < 8; ++e) {
        const int idx = tid * 8 + e;              // [j 0..31][c 0..63]
        lS[idx] = rf / cn[(idx >> 6) * 256 + s0 + (idx & 63)];
      }
    }

#pragma unroll
    for (int ct = 0; ct < 2; ++ct)
#pragma unroll
      for (int r = 0; r < 8; ++r) dsum[dir][ct][r] = 0.f;

    v8f acc0 = vzero;
    v8f acc1 = vzero;

    // Flattened pipeline over 31 negatives x 8 K-steps, double-buffered lK.
    const int NSTEP = 31 * 8;
    {  // prologue: copy for step 0
      const int j0 = (0 >= i) ? 1 : 0;
      cp16(&lK[0][ldrow * 32 + ldcol],
           Kb + ((size_t)(j0 * 256 + s0 + ldrow)) * 256 + ldcol);
    }
    for (int step = 0; step < NSTEP; ++step) {
      wait_async();        // copy for this step (issued at step-1) has landed
      __syncthreads();     // ... and everyone is done reading the other buffer
      if (step + 1 < NSTEP) {  // overlap next copy with this step's WMMAs
        const int jj = (step + 1) >> 3;
        const int jn = jj + ((jj >= i) ? 1 : 0);
        const int kn = ((step + 1) & 7) * 32;
        const unsigned short* src =
            Kb + ((size_t)(jn * 256 + s0 + ldrow)) * 256 + kn + ldcol;
        cp16(&lK[(step + 1) & 1][ldrow * 32 + ldcol], src);
        __builtin_prefetch(src + 32, 0, 1);  // hint for step+2
      }
      const int k0 = (step & 7) * 32;
      const unsigned short* bufK = lK[step & 1];
      // A fragment: lanes<16 take K 0..15, lanes>=16 take K 16..31.
      const uint4* qp = (const uint4*)&lQ[(m0 + lh) * 256 + k0];
      v16bf af = mkfrag(qp[hi * 2], qp[hi * 2 + 1]);
      {  // B fragment, C tile 0: lanes<16 K 0..15, lanes>=16 K 16..31
        const uint4* kp = (const uint4*)&bufK[(wcol * 32 + lh) * 32];
        v16bf bfr = mkfrag(kp[hi * 2], kp[hi * 2 + 1]);
        acc0 = __builtin_amdgcn_wmma_f32_16x16x32_bf16(false, af, false, bfr,
                                                       (short)0, acc0, false, false);
      }
      {  // B fragment, C tile 1
        const uint4* kp = (const uint4*)&bufK[(wcol * 32 + 16 + lh) * 32];
        v16bf bfr = mkfrag(kp[hi * 2], kp[hi * 2 + 1]);
        acc1 = __builtin_amdgcn_wmma_f32_16x16x32_bf16(false, af, false, bfr,
                                                       (short)0, acc1, false, false);
      }
      if ((step & 7) == 7) {  // j finished: scale, exp, accumulate, reset
        const int jj = step >> 3;
        const int j  = jj + ((jj >= i) ? 1 : 0);
        const float sc0 = lS[j * 64 + wcol * 32 + lh];
        const float sc1 = lS[j * 64 + wcol * 32 + 16 + lh];
#pragma unroll
        for (int r = 0; r < 8; ++r) {
          dsum[dir][0][r] += __expf(acc0[r] * sc0);
          dsum[dir][1][r] += __expf(acc1[r] * sc1);
        }
        acc0 = vzero;
        acc1 = vzero;
      }
    }
  }

  // out = -(log1p(down_V) + log1p(down_A)); C layout: lanes<16 M=r, >=16 M=8+r.
#pragma unroll
  for (int ct = 0; ct < 2; ++ct) {
    const int s = s0 + wcol * 32 + ct * 16 + lh;
#pragma unroll
    for (int r = 0; r < 8; ++r) {
      const int t = t0 + m0 + hi * 8 + r;
      out[((size_t)(i * 256 + t)) * 256 + s] =
          -(__logf(1.f + dsum[0][ct][r]) + __logf(1.f + dsum[1][ct][r]));
    }
  }
}

extern "C" void kernel_launch(void* const* d_in, const int* in_sizes, int n_in,
                              void* d_out, int out_size, void* d_ws, size_t ws_size,
                              hipStream_t stream) {
  (void)in_sizes; (void)n_in; (void)out_size; (void)ws_size;
  // setup_inputs order: pre_VF, pre_AF, back_VF, back_AF (pre_* are unused).
  const float* backV = (const float*)d_in[2];
  const float* backA = (const float*)d_in[3];

  // Workspace layout: bf16V [NH], bf16A [NH] (8.39MB), then norm floats (~64KB).
  unsigned short* bfV = (unsigned short*)d_ws;
  unsigned short* bfA = bfV + NH;
  float* nrm = (float*)(bfA + NH);
  float* out = (float*)d_out;

  convert_kernel<<<1024, 256, 0, stream>>>(backV, backA, bfV, bfA);
  norms_kernel<<<64, 256, 0, stream>>>(backV, backA, nrm);
  loss_kernel<<<512, 256, 0, stream>>>(bfV, bfA, nrm, out);
}
